// RelativisticGCN_16149077033547
// MI455X (gfx1250) — matrix operand order repaired and checked by reading, not compiled, not run on hardware
//
#include <hip/hip_runtime.h>
#include <hip/hip_bf16.h>
#include <math.h>

typedef __attribute__((ext_vector_type(16))) _Float16 v16h;
typedef __attribute__((ext_vector_type(8)))  float    v8f;

#define IN_CH   64
#define MAX_VEL 0.9f
#define EPS_W   1e-6f

// ---------------------------------------------------------------------------
// WMMA GEMM: H[M,N] = X[M,64] @ W[64,N] + bias[N]   (N = 64 or 32, M % 16 == 0)
// One wave computes one 16x16 output tile; K=64 -> two v_wmma_f32_16x16x32_f16.
// A fragment as loaded here (k order chosen by this code): lane L holds row M=(L&15),
//   halves j = K = ko+j (j=0..15), ko = (L<16?0:16)
// B fragment, same k order as A: lane L holds col N=(L&15), halves j = K = kb+j, kb = (L<16?0:16)
// C/D (f32, 8 VGPRs): lane L -> N=(L&15); vgpr r -> M = r + (L<16?0:8)
// ---------------------------------------------------------------------------
template <int N>
__global__ void __launch_bounds__(128)
rgcn_wmma_gemm(const float* __restrict__ X, const float* __restrict__ W,
               const float* __restrict__ bias, float* __restrict__ H, int M) {
    const int ntiles     = N / 16;
    const int tilesTotal = (M / 16) * ntiles;
    const int wave = blockIdx.x * (blockDim.x >> 5) + (threadIdx.x >> 5);
    const int lane = threadIdx.x & 31;
    if (wave >= tilesTotal) return;

    const int mtile = wave / ntiles;
    const int ntile = wave % ntiles;
    const int  l15  = lane & 15;
    const bool hi   = lane >= 16;

    const int m = mtile * 16 + l15;   // A row for this lane
    const int n = ntile * 16 + l15;   // B/C column for this lane

    v8f c = {};
#pragma unroll
    for (int kk = 0; kk < IN_CH; kk += 32) {
        // --- A fragment: 16 fp32 -> f16 converts per lane ---
        const int    ko = kk + (hi ? 16 : 0);
        const float* xr = X + (size_t)m * IN_CH + ko;
        v16h a;
#pragma unroll
        for (int j = 0; j < 8; ++j) {
            a[j]     = (_Float16)xr[j];        // K = ko + j
            a[8 + j] = (_Float16)xr[8 + j];    // K = ko + 8 + j
        }
        // --- B fragment: column n, 16 consecutive K rows ---
        const int    kb = kk + (hi ? 16 : 0);
        const float* wr = W + (size_t)kb * N + n;
        v16h b;
#pragma unroll
        for (int j = 0; j < 16; ++j)
            b[j] = (_Float16)wr[(size_t)j * N];

        c = __builtin_amdgcn_wmma_f32_16x16x32_f16(
                /*neg_a=*/false, a, /*neg_b=*/false, b,
                /*c_mod=*/(short)0, c, /*reuse_a=*/false, /*reuse_b=*/false);
    }

    const float bn = bias[n];
#pragma unroll
    for (int r = 0; r < 8; ++r) {
        const int mr = mtile * 16 + r + (hi ? 8 : 0);
        H[(size_t)mr * N + n] = c[r] + bn;
    }
}

// ---------------------------------------------------------------------------
// Fused edge kernel: one wave32 per edge.
//   w = sqrt(1 - (MAX_VEL * tanh(||feat[src]-feat[dst]||))^2 + eps)
//   agg[dst] += h[src] * w  (f32 global atomics, L2-resident accumulators)
//   deg[dst] += 1 (layer 1 only)
// feat always has 64 channels (2 per lane); messages have CH_MSG channels.
// ---------------------------------------------------------------------------
template <int CH_MSG, bool COUNT_DEG>
__global__ void __launch_bounds__(256)
rgcn_edge_scatter(const float* __restrict__ feat, const float* __restrict__ msg,
                  const int* __restrict__ src, const int* __restrict__ dst,
                  float* __restrict__ agg, float* __restrict__ deg, int n_edges) {
    const int wave = blockIdx.x * (blockDim.x >> 5) + (threadIdx.x >> 5);
    const int lane = threadIdx.x & 31;
    if (wave >= n_edges) return;

    const int s = src[wave];
    const int d = dst[wave];

    // squared-distance partial: 2 channels per lane
    const float2 fa = ((const float2*)(feat + (size_t)s * IN_CH))[lane];
    const float2 fb = ((const float2*)(feat + (size_t)d * IN_CH))[lane];
    const float dx = fa.x - fb.x, dy = fa.y - fb.y;
    float p = dx * dx + dy * dy;
#pragma unroll
    for (int off = 16; off > 0; off >>= 1)
        p += __shfl_xor(p, off, 32);          // wave32 butterfly reduction

    const float nrm = sqrtf(p);
    const float v   = tanhf(nrm) * MAX_VEL;
    const float w   = sqrtf(1.0f - v * v + EPS_W);   // 1/gamma

    if (CH_MSG == 64) {
        const float2 mv = ((const float2*)(msg + (size_t)s * 64))[lane];
        float* ap = agg + (size_t)d * 64 + lane * 2;
        atomicAdd(ap,     mv.x * w);
        atomicAdd(ap + 1, mv.y * w);
    } else {  // 32 channels: one per lane
        const float mv = msg[(size_t)s * 32 + lane];
        atomicAdd(agg + (size_t)d * 32 + lane, mv * w);
    }
    if (COUNT_DEG && lane == 0) atomicAdd(deg + d, 1.0f);
}

// ---------------------------------------------------------------------------
// out[i] = (relu?) agg[i] / max(deg[node], 1)      (in-place safe)
// ---------------------------------------------------------------------------
template <int CH, bool RELU>
__global__ void __launch_bounds__(256)
rgcn_norm(const float* __restrict__ agg, const float* __restrict__ deg,
          float* __restrict__ out, int n_nodes) {
    const size_t i = (size_t)blockIdx.x * blockDim.x + threadIdx.x;
    if (i >= (size_t)n_nodes * CH) return;
    const int   node = (int)(i / CH);
    const float dv   = fmaxf(deg[node], 1.0f);
    float val = agg[i] / dv;
    if (RELU) val = fmaxf(val, 0.0f);
    out[i] = val;
}

// ---------------------------------------------------------------------------
extern "C" void kernel_launch(void* const* d_in, const int* in_sizes, int n_in,
                              void* d_out, int out_size, void* d_ws, size_t ws_size,
                              hipStream_t stream) {
    const float* x  = (const float*)d_in[0];
    const int*   ei = (const int*)d_in[1];     // [2, E] row-major: src then dst
    const float* W1 = (const float*)d_in[2];
    const float* b1 = (const float*)d_in[3];
    const float* W2 = (const float*)d_in[4];
    const float* b2 = (const float*)d_in[5];

    const int NN = in_sizes[0] / IN_CH;        // 100000
    const int E  = in_sizes[1] / 2;            // 1600000
    const int* src = ei;
    const int* dst = ei + E;

    // workspace layout (buffers reused across stages):
    //   buf0: h1, then out1 (relu'd)   [NN*64]
    //   buf1: agg1, then h2            [NN*64]
    //   deg                            [NN]
    float* buf0 = (float*)d_ws;
    float* buf1 = buf0 + (size_t)NN * 64;
    float* deg  = buf1 + (size_t)NN * 64;
    float* outp = (float*)d_out;               // doubles as agg2

    hipMemsetAsync(buf1, 0, (size_t)NN * 64 * sizeof(float), stream);
    hipMemsetAsync(deg,  0, (size_t)NN * sizeof(float), stream);
    hipMemsetAsync(outp, 0, (size_t)NN * 32 * sizeof(float), stream);

    const int edgeBlocks = (E + 7) / 8;        // 8 waves (edges) per 256-thr block

    // ---- Layer 1 ----
    {
        const int tiles  = (NN / 16) * (64 / 16);          // 25000 waves
        const int blocks = (tiles + 3) / 4;                // 4 waves per block
        rgcn_wmma_gemm<64><<<blocks, 128, 0, stream>>>(x, W1, b1, buf0, NN);
    }
    rgcn_edge_scatter<64, true><<<edgeBlocks, 256, 0, stream>>>(
        x, buf0, src, dst, buf1, deg, E);
    rgcn_norm<64, true><<<(NN * 64 + 255) / 256, 256, 0, stream>>>(
        buf1, deg, buf0, NN);                              // out1 -> buf0

    // ---- Layer 2 ----
    {
        const int tiles  = (NN / 16) * (32 / 16);          // 12500 waves
        const int blocks = (tiles + 3) / 4;
        rgcn_wmma_gemm<32><<<blocks, 128, 0, stream>>>(buf0, W2, b2, buf1, NN);
    }
    rgcn_edge_scatter<32, false><<<edgeBlocks, 256, 0, stream>>>(
        buf0, buf1, src, dst, outp, nullptr, E);
    rgcn_norm<32, false><<<(NN * 32 + 255) / 256, 256, 0, stream>>>(
        outp, deg, outp, NN);
}
